// CrossCostVolume_42219528520108
// MI455X (gfx1250) — matrix operand order repaired and checked by reading, not compiled, not run on hardware
//
#include <hip/hip_runtime.h>
#include <math.h>

// ---------------------------------------------------------------------------
// CDNA5 / gfx1250 implementation of the CrossCostVolume reference.
// Heavy pairwise MLP (1024x1024 pairs, 128->64->64) runs on
// v_wmma_f32_16x16x32_bf16 with f32 accumulation; BN(training) stats force a
// 3-pass recompute schedule (stats-y1, stats-y2, final). All reductions are
// deterministic (no float atomics). Inner loop uses packed bf16 VALU
// (v_pk_add_bf16 / v_pk_max_num_bf16), weights are DMA'd to LDS with
// global_load_async_to_lds_b32 (ASYNCcnt), LDS transposes use s_wait_dscnt.
// ---------------------------------------------------------------------------

#define DEVI __device__ __forceinline__

typedef __attribute__((ext_vector_type(16))) __bf16 v16bf;
typedef __attribute__((ext_vector_type(8)))  float  v8f;
typedef __attribute__((ext_vector_type(4)))  unsigned uv4;

union AF { v16bf v; unsigned short h[16]; unsigned u[8]; uv4 q[2]; };

DEVI float bf2f(unsigned short s) { return __uint_as_float(((unsigned)s) << 16); }
DEVI unsigned short f2bf(float x) {
  unsigned u = __float_as_uint(x);
  return (unsigned short)((u + 0x7FFFu + ((u >> 16) & 1u)) >> 16);  // RNE
}

// packed 2xbf16 add / relu on CDNA5 VOP3P
DEVI unsigned pk_add_bf16(unsigned a, unsigned b) {
  unsigned d;
  asm("v_pk_add_bf16 %0, %1, %2" : "=v"(d) : "v"(a), "v"(b));
  return d;
}
DEVI unsigned pk_relu_bf16(unsigned a) {
  unsigned d;
  asm("v_pk_max_num_bf16 %0, %1, 0" : "=v"(d) : "v"(a));
  return d;
}

DEVI v8f wmma_bf16(const AF& a, const AF& b, v8f c) {
  return __builtin_amdgcn_wmma_f32_16x16x32_bf16(false, a.v, false, b.v,
                                                 (short)0, c, false, false);
}

// B fragment: 32x16 (KxN) tile of row-major [K x ld] bf16. lane L (n=n0+(L&15),
// hi=L>>4): elems 0..7 = K=k0+8*hi.., elems 8..15 = K=k0+16+8*hi.. (strided).
DEVI void loadB(AF& f, const unsigned short* W, int ld, int k0, int n0, int lane) {
  const unsigned short* p = W + (size_t)(k0 + 8 * (lane >> 4)) * ld + n0 + (lane & 15);
#pragma unroll
  for (int s = 0; s < 8; s++) {
    f.h[s] = p[(size_t)s * ld];
    f.h[8 + s] = p[(size_t)(16 + s) * ld];
  }
}

// A fragment: 16x32 (MxK) tile at k0 of row-major [16 x ld] bf16.
// lane L (m=L&15, hi=L>>4): elems 0..7 = K=k0+8*hi.., elems 8..15 = K=k0+16+8*hi..
DEVI void loadA(AF& f, const unsigned short* X, int ld, int k0, int lane) {
  int m = lane & 15, hi = lane >> 4;
  const unsigned short* p = X + m * ld + k0 + 8 * hi;
  f.q[0] = *(const uv4*)p;
  f.q[1] = *(const uv4*)(p + 16);
}

// ------------------------- small prep kernels ------------------------------

// rows 0..1023: pi_n (+ wxyz), rows 1024..2047: qi_n (+ bf16 transpose for B)
__global__ __launch_bounds__(256) void k_prep(
    const float* wx, const float* wp, const float* fx, const float* fp,
    const float* lz, float* wxyz, float* pin, float* qin,
    unsigned short* pinb, unsigned short* qiTb) {
  int row = blockIdx.x * 8 + (threadIdx.x >> 5);
  int lane = threadIdx.x & 31;
  bool isQ = row >= 1024;
  int n = isQ ? row - 1024 : row;
  const float* src = isQ ? fp : wp;
  float x0 = src[n * 64 + lane];
  float x1 = src[n * 64 + lane + 32];
  float ss = x0 * x0 + x1 * x1;
#pragma unroll
  for (int s = 16; s > 0; s >>= 1) ss += __shfl_xor(ss, s, 32);
  float scale = 1.f / fmaxf(sqrtf(ss), 1e-12f);
  float y0 = x0 * scale, y1 = x1 * scale;
  if (!isQ) {
    pin[n * 64 + lane] = y0; pin[n * 64 + lane + 32] = y1;
    pinb[n * 64 + lane] = f2bf(y0); pinb[n * 64 + lane + 32] = f2bf(y1);
    if (lane < 3) wxyz[n * 3 + lane] = wx[n * 3 + lane] * lz[n];
  } else {
    qin[n * 64 + lane] = y0; qin[n * 64 + lane + 32] = y1;
    qiTb[(size_t)lane * 1024 + n] = f2bf(y0);
    qiTb[(size_t)(lane + 32) * 1024 + n] = f2bf(y1);
  }
}

__global__ void k_f2bf(const float* s, unsigned short* d, int n) {
  int i = blockIdx.x * 256 + threadIdx.x;
  if (i < n) d[i] = f2bf(s[i]);
}

// hp = wxyz@W0[0:3] + pi_n@W0[6:70] ; hq = f2_xyz@W0[3:6] + qi_n@W0[70:134]
__global__ __launch_bounds__(256) void k_hphq(
    const float* W0, const float* wxyz, const float* pin, const float* qin,
    const float* fx, float* hp, float* hq) {
  int n = blockIdx.x;
  int t = threadIdx.x;
  int c = t & 127;
  float acc = 0.f;
  if (t < 128) {
#pragma unroll
    for (int d = 0; d < 3; d++) acc = fmaf(wxyz[n * 3 + d], W0[d * 128 + c], acc);
    for (int k = 0; k < 64; k++) acc = fmaf(pin[n * 64 + k], W0[(6 + k) * 128 + c], acc);
    hp[n * 128 + c] = acc;
  } else {
#pragma unroll
    for (int d = 0; d < 3; d++) acc = fmaf(fx[n * 3 + d], W0[(3 + d) * 128 + c], acc);
    for (int k = 0; k < 64; k++) acc = fmaf(qin[n * 64 + k], W0[(70 + k) * 128 + c], acc);
    hq[n * 128 + c] = acc;
  }
}

// BN1 stats are analytic: var over (n,m) of hp[n]+hq[m] = var(hp)+var(hq).
__global__ void k_bn1_stats(const float* hp, const float* hq, const float* gamma0,
                            float* a1, float* mup, float* muq) {
  int c = threadIdx.x;  // 128 threads
  float sp = 0, qp = 0, sq = 0, qq = 0;
  for (int n = 0; n < 1024; n++) {
    float v = hp[n * 128 + c]; sp += v; qp += v * v;
    v = hq[n * 128 + c]; sq += v; qq += v * v;
  }
  float mp = sp * (1.f / 1024.f), mq = sq * (1.f / 1024.f);
  float vp = qp * (1.f / 1024.f) - mp * mp;
  float vq = qq * (1.f / 1024.f) - mq * mq;
  a1[c] = gamma0[c] * rsqrtf(vp + vq + 1e-5f);
  mup[c] = mp; muq[c] = mq;
}

// Fold BN1 into bf16 row arrays: x1 = relu(hp2[n] + hq2[m]).
__global__ void k_fold(const float* hp, const float* hq, const float* a1,
                       const float* mup, const float* muq, const float* beta0,
                       unsigned short* hp2b, unsigned short* hq2b) {
  int i = blockIdx.x * 256 + threadIdx.x;  // 131072
  int c = i & 127;
  hp2b[i] = f2bf(a1[c] * (hp[i] - mup[c]));
  hq2b[i] = f2bf(fmaf(a1[c], hq[i] - muq[c], beta0[c]));
}

// sim = pi_n @ qi_n^T via WMMA bf16, 16x16 tiles.
__global__ __launch_bounds__(256) void k_sim(const unsigned short* pinb,
                                             const unsigned short* qiTb,
                                             float* sim) {
  int tile = blockIdx.x * 8 + (threadIdx.x >> 5);
  int lane = threadIdx.x & 31;
  int tn = tile >> 6, tm = tile & 63;
  v8f acc;
#pragma unroll
  for (int r = 0; r < 8; r++) acc[r] = 0.f;
#pragma unroll
  for (int kc = 0; kc < 2; kc++) {
    AF a, b;
    loadA(a, pinb + (size_t)tn * 16 * 64, 64, kc * 32, lane);
    loadB(b, qiTb, 1024, kc * 32, tm * 16, lane);
    acc = wmma_bf16(a, b, acc);
  }
  int m16 = lane & 15, hi = lane >> 4;
#pragma unroll
  for (int r = 0; r < 8; r++)
    sim[(size_t)(tn * 16 + r + 8 * hi) * 1024 + tm * 16 + m16] = acc[r];
}

__global__ __launch_bounds__(256) void k_rowmax(const float* sim, float* rmax) {
  int row = blockIdx.x * 8 + (threadIdx.x >> 5), lane = threadIdx.x & 31;
  float m = -3.4e38f;
  for (int j = lane; j < 1024; j += 32) m = fmaxf(m, sim[(size_t)row * 1024 + j]);
#pragma unroll
  for (int s = 16; s > 0; s >>= 1) m = fmaxf(m, __shfl_xor(m, s, 32));
  if (lane == 0) rmax[row] = m;
}

// stores 1/(colmax + 1e-10) so pass C multiplies instead of divides
__global__ void k_colmax(const float* sim, float* cinv) {
  int c = blockIdx.x * 256 + threadIdx.x;
  float m = -3.4e38f;
  for (int n = 0; n < 1024; n++) m = fmaxf(m, sim[(size_t)n * 1024 + c]);
  cinv[c] = 1.f / (m + 1e-10f);
}

// ------------------- the big pairwise-MLP WMMA pass ------------------------
// MODE 0: GEMM1 only, emit per-n partial (sum,sumsq) of y1 (64 ch)
// MODE 1: GEMM1+BN2+relu+GEMM2, emit partial stats of y2
// MODE 2: full chain + BN3 + relu + sim_rate + max over m -> pi_feat
template <int MODE>
__global__ __launch_bounds__(256) void k_pair_mlp(
    const unsigned short* __restrict__ hp2b, const unsigned short* __restrict__ hq2b,
    const unsigned short* __restrict__ w1b, const unsigned short* __restrict__ w2b,
    const float* __restrict__ b1, const float* __restrict__ a2v,
    const float* __restrict__ o2v, const float* __restrict__ b2,
    const float* __restrict__ a3v, const float* __restrict__ o3v,
    const float* __restrict__ sim, const float* __restrict__ rmax,
    const float* __restrict__ cinv, float* __restrict__ pS,
    float* __restrict__ pQ, float* __restrict__ pi_feat) {
  __shared__ unsigned short sW1[128 * 72];
  __shared__ unsigned short sW2[64 * 72];
  __shared__ unsigned short sX2[8][16 * 72];
  // DMA weights into padded LDS rows with CDNA5 async loads (ASYNCcnt).
  for (int i = threadIdx.x; i < 128 * 32; i += 256) {  // dwords, 32 dw/row
    unsigned lds = (unsigned)(size_t)(void*)&sW1[(i >> 5) * 72 + (i & 31) * 2];
    asm volatile("global_load_async_to_lds_b32 %0, %1, %2"
                 :: "v"(lds), "v"(i * 4), "s"(w1b) : "memory");
  }
  if (MODE >= 1) {
    for (int i = threadIdx.x; i < 64 * 32; i += 256) {
      unsigned lds = (unsigned)(size_t)(void*)&sW2[(i >> 5) * 72 + (i & 31) * 2];
      asm volatile("global_load_async_to_lds_b32 %0, %1, %2"
                   :: "v"(lds), "v"(i * 4), "s"(w2b) : "memory");
    }
  }
  asm volatile("s_wait_asynccnt 0x0" ::: "memory");
  __syncthreads();

  int wave = threadIdx.x >> 5, lane = threadIdx.x & 31;
  int n = blockIdx.x * 8 + wave;
  int m16 = lane & 15, hi = lane >> 4;
  const unsigned short* hprow = hp2b + (size_t)n * 128;
  unsigned short* X2 = sX2[wave];

  float bias1[4], bias2[4] = {0, 0, 0, 0};
  float sA2[4] = {0, 0, 0, 0}, sO2[4] = {0, 0, 0, 0};
  float sA3[4] = {0, 0, 0, 0}, sO3[4] = {0, 0, 0, 0};
#pragma unroll
  for (int t = 0; t < 4; t++) {
    int c = 16 * t + m16;
    bias1[t] = b1[c];
    if (MODE >= 1) { bias2[t] = b2[c]; sA2[t] = a2v[c]; sO2[t] = o2v[c]; }
    if (MODE == 2) { sA3[t] = a3v[c]; sO3[t] = o3v[c]; }
  }
  float accS[4] = {0, 0, 0, 0}, accQ[4] = {0, 0, 0, 0};
  float vmax[4];
#pragma unroll
  for (int t = 0; t < 4; t++) vmax[t] = -3.4e38f;
  float inv_rm = (MODE == 2) ? 1.f / (rmax[n] + 1e-10f) : 0.f;

  // hp fragments are invariant over the m loop: hoist into registers.
  unsigned HP[4][8];
#pragma unroll
  for (int kc = 0; kc < 4; kc++) {
    int c0 = kc * 32 + 8 * hi;
    union { uv4 q; unsigned u[4]; } P0, P1;
    P0.q = *(const uv4*)(hprow + c0);
    P1.q = *(const uv4*)(hprow + c0 + 16);
#pragma unroll
    for (int j = 0; j < 4; j++) { HP[kc][j] = P0.u[j]; HP[kc][4 + j] = P1.u[j]; }
  }

  for (int mb = 0; mb < 64; mb++) {
    const unsigned short* hqrow = hq2b + (size_t)(mb * 16 + m16) * 128;
    if (mb < 63)
      __builtin_prefetch(hq2b + (size_t)((mb + 1) * 16 + m16) * 128, 0, 3);
    v8f y1[4];
#pragma unroll
    for (int t = 0; t < 4; t++) {
#pragma unroll
      for (int r = 0; r < 8; r++) y1[t][r] = bias1[t];
    }
#pragma unroll
    for (int kc = 0; kc < 4; kc++) {
      AF a;
      {
        int c0 = kc * 32 + 8 * hi;
        union { uv4 q; unsigned u[4]; } Q0, Q1;
        Q0.q = *(const uv4*)(hqrow + c0);
        Q1.q = *(const uv4*)(hqrow + c0 + 16);
#pragma unroll
        for (int j = 0; j < 4; j++) {
          a.u[j]     = pk_relu_bf16(pk_add_bf16(HP[kc][j],     Q0.u[j]));
          a.u[4 + j] = pk_relu_bf16(pk_add_bf16(HP[kc][4 + j], Q1.u[j]));
        }
      }
#pragma unroll
      for (int t = 0; t < 4; t++) {
        AF b;
        loadB(b, sW1, 72, kc * 32, t * 16, lane);
        y1[t] = wmma_bf16(a, b, y1[t]);
      }
    }

    if (MODE == 0) {
#pragma unroll
      for (int t = 0; t < 4; t++) {
#pragma unroll
        for (int r = 0; r < 8; r++) {
          float v = y1[t][r];
          accS[t] += v; accQ[t] += v * v;
        }
      }
    } else {
      // BN2 + relu, transpose to A layout via per-wave LDS staging
#pragma unroll
      for (int t = 0; t < 4; t++) {
        int c = 16 * t + m16;
#pragma unroll
        for (int r = 0; r < 8; r++) {
          float v = fmaxf(fmaf(sA2[t], y1[t][r], sO2[t]), 0.f);
          X2[(r + 8 * hi) * 72 + c] = f2bf(v);
        }
      }
      asm volatile("s_wait_dscnt 0x0" ::: "memory");
      v8f y2[4];
#pragma unroll
      for (int t = 0; t < 4; t++) {
#pragma unroll
        for (int r = 0; r < 8; r++) y2[t][r] = bias2[t];
      }
#pragma unroll
      for (int kc = 0; kc < 2; kc++) {
        AF af;
        loadA(af, X2, 72, kc * 32, lane);
#pragma unroll
        for (int t = 0; t < 4; t++) {
          AF b;
          loadB(b, sW2, 72, kc * 32, t * 16, lane);
          y2[t] = wmma_bf16(af, b, y2[t]);
        }
      }
      if (MODE == 1) {
#pragma unroll
        for (int t = 0; t < 4; t++) {
#pragma unroll
          for (int r = 0; r < 8; r++) {
            float v = y2[t][r];
            accS[t] += v; accQ[t] += v * v;
          }
        }
      } else {
        float sr[8];
#pragma unroll
        for (int r = 0; r < 8; r++) {
          int m = mb * 16 + r + 8 * hi;
          float s = sim[(size_t)n * 1024 + m];
          sr[r] = s * s * inv_rm * cinv[m];
        }
#pragma unroll
        for (int t = 0; t < 4; t++) {
#pragma unroll
          for (int r = 0; r < 8; r++) {
            float h = fmaxf(fmaf(sA3[t], y2[t][r], sO3[t]), 0.f);
            vmax[t] = fmaxf(vmax[t], h * sr[r]);
          }
        }
      }
    }
  }

  if (MODE <= 1) {
#pragma unroll
    for (int t = 0; t < 4; t++) {
      float s = accS[t] + __shfl_xor(accS[t], 16, 32);
      float q = accQ[t] + __shfl_xor(accQ[t], 16, 32);
      if (hi == 0) {
        pS[(size_t)n * 64 + 16 * t + m16] = s;
        pQ[(size_t)n * 64 + 16 * t + m16] = q;
      }
    }
  } else {
#pragma unroll
    for (int t = 0; t < 4; t++) {
      float v = fmaxf(vmax[t], __shfl_xor(vmax[t], 16, 32));
      if (hi == 0) pi_feat[(size_t)n * 64 + 16 * t + m16] = v;
    }
  }
}

// ------------------- deterministic BN stats reductions ---------------------
__global__ __launch_bounds__(256) void k_stats1(const float* x, int rows, int ch,
                                                float* pS, float* pQ) {
  int c = threadIdx.x % ch;
  int rpb = 256 / ch;
  int sub = threadIdx.x / ch;
  int r0 = blockIdx.x * rpb + sub;
  int stride = gridDim.x * rpb;
  float s = 0, q = 0;
  for (int r = r0; r < rows; r += stride) {
    float v = x[(size_t)r * ch + c];
    s += v; q += v * v;
  }
  pS[(size_t)(blockIdx.x * rpb + sub) * ch + c] = s;
  pQ[(size_t)(blockIdx.x * rpb + sub) * ch + c] = q;
}

__global__ void k_reduce_cols(const float* p, int rows, int ch, float* out) {
  int c = threadIdx.x;
  if (c >= ch) return;
  float s = 0;
  for (int r = 0; r < rows; r++) s += p[(size_t)r * ch + c];
  out[c] = s;
}

__global__ void k_finalize_ao(const float* sum, const float* sq, float cnt,
                              const float* gamma, const float* beta, float* a,
                              float* o, int ch) {
  int c = blockIdx.x * 256 + threadIdx.x;
  if (c >= ch) return;
  float mu = sum[c] / cnt;
  float var = sq[c] / cnt - mu * mu;
  float aa = gamma[c] * rsqrtf(var + 1e-5f);
  a[c] = aa;
  o[c] = beta[c] - mu * aa;
}

// ----------------------------- KNN (K=16) ----------------------------------
__global__ __launch_bounds__(256) void k_knn(const float* wxyz, int* idx) {
  int wave = threadIdx.x >> 5, lane = threadIdx.x & 31;
  int n = blockIdx.x * 8 + wave;
  float px = wxyz[n * 3], py = wxyz[n * 3 + 1], pz = wxyz[n * 3 + 2];
  float d[32];
#pragma unroll
  for (int j = 0; j < 32; j++) {
    int m = lane * 32 + j;
    float dx = wxyz[m * 3] - px, dy = wxyz[m * 3 + 1] - py, dz = wxyz[m * 3 + 2] - pz;
    d[j] = dx * dx + dy * dy + dz * dz;
  }
  for (int it = 0; it < 16; it++) {
    float bv = 3.4e38f;
    int bm = 0x7FFFFFFF;
#pragma unroll
    for (int j = 0; j < 32; j++) {
      if (d[j] < bv) { bv = d[j]; bm = lane * 32 + j; }
    }
#pragma unroll
    for (int s = 16; s > 0; s >>= 1) {
      float ov = __shfl_xor(bv, s, 32);
      int om = __shfl_xor(bm, s, 32);
      if (ov < bv || (ov == bv && om < bm)) { bv = ov; bm = om; }
    }
    if (lane == 0) idx[n * 16 + it] = bm;
    if ((bm >> 5) == lane) {
#pragma unroll
      for (int j = 0; j < 32; j++)
        if ((bm & 31) == j) d[j] = 3.4e38f;
    }
  }
}

// pc_enc: 10 -> 64 (tiny, plain VALU), pre-BN output to ws
__global__ __launch_bounds__(256) void k_enc(const float* wxyz, const int* idx,
                                             const float* Wenc, const float* benc,
                                             float* out) {
  int t = blockIdx.x * 256 + threadIdx.x;  // 16384
  int n = t >> 4;
  int m = idx[t];
  float in[10];
  in[0] = wxyz[n * 3]; in[1] = wxyz[n * 3 + 1]; in[2] = wxyz[n * 3 + 2];
  in[3] = wxyz[m * 3]; in[4] = wxyz[m * 3 + 1]; in[5] = wxyz[m * 3 + 2];
  in[6] = in[3] - in[0]; in[7] = in[4] - in[1]; in[8] = in[5] - in[2];
  in[9] = sqrtf(in[6] * in[6] + in[7] * in[7] + in[8] * in[8] + 1e-20f);
  for (int c = 0; c < 64; c++) {
    float acc = benc[c];
#pragma unroll
    for (int i = 0; i < 10; i++) acc = fmaf(in[i], Wenc[i * 64 + c], acc);
    out[(size_t)t * 64 + c] = acc;
  }
}

// mlp2 layer 0: [enc(64)|warped_points(64)|gfeat(64)] (16 neighbors) @ W(192x128)
__global__ __launch_bounds__(256) void k_mlp2a(
    const float* wsenc, const float* aE, const float* oE, const float* wpoints,
    const float* pifeat, const int* idx, const unsigned short* w2ab,
    const float* bA, float* wsa) {
  __shared__ unsigned short sX[8][16 * 200];
  int wave = threadIdx.x >> 5, lane = threadIdx.x & 31;
  int n = blockIdx.x * 8 + wave;
  int m16 = lane & 15, hi = lane >> 4;
  unsigned short* X = sX[wave];
  for (int r = 0; r < 16; r++) {
    int gi = idx[n * 16 + r];
#pragma unroll
    for (int seg = 0; seg < 6; seg++) {
      int c = seg * 32 + lane;
      float v;
      if (c < 64)
        v = fmaxf(fmaf(aE[c], wsenc[(size_t)(n * 16 + r) * 64 + c], oE[c]), 0.f);
      else if (c < 128)
        v = wpoints[n * 64 + c - 64];
      else
        v = pifeat[(size_t)gi * 64 + c - 128];
      X[r * 200 + c] = f2bf(v);
    }
  }
  asm volatile("s_wait_dscnt 0x0" ::: "memory");
  v8f y[8];
#pragma unroll
  for (int t = 0; t < 8; t++) {
    float b = bA[16 * t + m16];
#pragma unroll
    for (int r = 0; r < 8; r++) y[t][r] = b;
  }
  for (int kc = 0; kc < 6; kc++) {
    AF a;
    loadA(a, X, 200, kc * 32, lane);
#pragma unroll
    for (int t = 0; t < 8; t++) {
      AF b;
      loadB(b, w2ab, 128, kc * 32, t * 16, lane);
      y[t] = wmma_bf16(a, b, y[t]);
    }
  }
#pragma unroll
  for (int t = 0; t < 8; t++) {
    int c = 16 * t + m16;
#pragma unroll
    for (int r = 0; r < 8; r++)
      wsa[(size_t)(n * 16 + r + 8 * hi) * 128 + c] = y[t][r];
  }
}

// mlp2 layer 1: relu(bn(wsa)) (16x128) @ W(128x64)
__global__ __launch_bounds__(256) void k_mlp2b(
    const float* wsa, const float* aM, const float* oM,
    const unsigned short* w2bb, const float* bB, float* wsb) {
  __shared__ unsigned short sX[8][16 * 136];
  int wave = threadIdx.x >> 5, lane = threadIdx.x & 31;
  int n = blockIdx.x * 8 + wave;
  int m16 = lane & 15, hi = lane >> 4;
  unsigned short* X = sX[wave];
  for (int r = 0; r < 16; r++) {
#pragma unroll
    for (int seg = 0; seg < 4; seg++) {
      int c = seg * 32 + lane;
      float v = fmaxf(fmaf(aM[c], wsa[(size_t)(n * 16 + r) * 128 + c], oM[c]), 0.f);
      X[r * 136 + c] = f2bf(v);
    }
  }
  asm volatile("s_wait_dscnt 0x0" ::: "memory");
  v8f y[4];
#pragma unroll
  for (int t = 0; t < 4; t++) {
    float b = bB[16 * t + m16];
#pragma unroll
    for (int r = 0; r < 8; r++) y[t][r] = b;
  }
  for (int kc = 0; kc < 4; kc++) {
    AF a;
    loadA(a, X, 136, kc * 32, lane);
#pragma unroll
    for (int t = 0; t < 4; t++) {
      AF b;
      loadB(b, w2bb, 64, kc * 32, t * 16, lane);
      y[t] = wmma_bf16(a, b, y[t]);
    }
  }
#pragma unroll
  for (int t = 0; t < 4; t++) {
    int c = 16 * t + m16;
#pragma unroll
    for (int r = 0; r < 8; r++)
      wsb[(size_t)(n * 16 + r + 8 * hi) * 64 + c] = y[t][r];
  }
}

// final: relu(bn(wsb)), softmax over 16 neighbors, weighted sum of gfeat
__global__ __launch_bounds__(256) void k_softmax_out(
    const float* wsb, const float* aM, const float* oM, const int* idx,
    const float* pifeat, float* out) {
  int wave = threadIdx.x >> 5, lane = threadIdx.x & 31;
  int n = blockIdx.x * 8 + wave;
#pragma unroll
  for (int half = 0; half < 2; half++) {
    int c = lane + 32 * half;
    float aa = aM[c], oo = oM[c];
    float z[16];
    float mx = -3.4e38f;
#pragma unroll
    for (int k = 0; k < 16; k++) {
      z[k] = fmaxf(fmaf(aa, wsb[(size_t)(n * 16 + k) * 64 + c], oo), 0.f);
      mx = fmaxf(mx, z[k]);
    }
    float s = 0.f;
#pragma unroll
    for (int k = 0; k < 16; k++) { z[k] = __expf(z[k] - mx); s += z[k]; }
    float inv = 1.f / s;
    float acc = 0.f;
#pragma unroll
    for (int k = 0; k < 16; k++)
      acc += z[k] * inv * pifeat[(size_t)idx[n * 16 + k] * 64 + c];
    out[(size_t)n * 64 + c] = acc;
  }
}

// ---------------------------------------------------------------------------

extern "C" void kernel_launch(void* const* d_in, const int* in_sizes, int n_in,
                              void* d_out, int out_size, void* d_ws,
                              size_t ws_size, hipStream_t stream) {
  (void)in_sizes; (void)n_in; (void)out_size; (void)ws_size;
  const float* warped_xyz    = (const float*)d_in[0];
  const float* warped_points = (const float*)d_in[1];
  const float* f2_xyz        = (const float*)d_in[2];
  const float* f2_points     = (const float*)d_in[3];
  const float* lidar_z       = (const float*)d_in[4];
  // params flattened in setup_inputs() insertion order: mlp1[0..2], pc_enc,
  // mlp2[0..1]; each conv dict as (W, b, gamma, beta).
  const float *W0 = (const float*)d_in[5],  *b0 = (const float*)d_in[6],
              *g0 = (const float*)d_in[7],  *be0 = (const float*)d_in[8];
  const float *W1 = (const float*)d_in[9],  *b1 = (const float*)d_in[10],
              *g1 = (const float*)d_in[11], *be1 = (const float*)d_in[12];
  const float *W2 = (const float*)d_in[13], *b2 = (const float*)d_in[14],
              *g2 = (const float*)d_in[15], *be2 = (const float*)d_in[16];
  const float *We = (const float*)d_in[17], *bE = (const float*)d_in[18],
              *gE = (const float*)d_in[19], *beE = (const float*)d_in[20];
  const float *WA = (const float*)d_in[21], *bA = (const float*)d_in[22],
              *gA = (const float*)d_in[23], *beA = (const float*)d_in[24];
  const float *WB = (const float*)d_in[25], *bB = (const float*)d_in[26],
              *gB = (const float*)d_in[27], *beB = (const float*)d_in[28];
  (void)b0;

  float* base = (float*)d_ws;
  size_t off = 0;
  auto F = [&](size_t nf) { float* p = base + off; off += nf; return p; };

  float* wxyz   = F(1024 * 3);
  float* pin    = F(1024 * 64);
  float* qin    = F(1024 * 64);
  float* hp     = F(1024 * 128);
  float* hq     = F(1024 * 128);
  float* sim    = F(1024 * 1024);
  float* rmax   = F(1024);
  float* cinv   = F(1024);
  float* pifeat = F(1024 * 64);
  float* wsenc  = F(16384 * 64);
  float* wsa    = F(16384 * 128);
  float* wsb    = F(16384 * 64);
  unsigned short* pinb = (unsigned short*)F(1024 * 64 / 2);
  unsigned short* qiTb = (unsigned short*)F(1024 * 64 / 2);
  unsigned short* hp2b = (unsigned short*)F(1024 * 128 / 2);
  unsigned short* hq2b = (unsigned short*)F(1024 * 128 / 2);
  unsigned short* w1b  = (unsigned short*)F(128 * 64 / 2);
  unsigned short* w2b  = (unsigned short*)F(64 * 64 / 2);
  unsigned short* w2ab = (unsigned short*)F(192 * 128 / 2);
  unsigned short* w2bb = (unsigned short*)F(128 * 64 / 2);
  int* idx = (int*)F(1024 * 16);
  float* a1 = F(128); float* mup = F(128); float* muq = F(128);
  float* partS = F(65536); float* partQ = F(65536);
  float* st2S = F(64);  float* st2Q = F(64);
  float* st3S = F(64);  float* st3Q = F(64);
  float* stES = F(64);  float* stEQ = F(64);
  float* stAS = F(128); float* stAQ = F(128);
  float* stBS = F(64);  float* stBQ = F(64);
  float* a2 = F(64);  float* o2 = F(64);
  float* a3 = F(64);  float* o3 = F(64);
  float* aE = F(64);  float* oE = F(64);
  float* aM1 = F(128); float* oM1 = F(128);
  float* aM2 = F(64);  float* oM2 = F(64);

  // ---- front end ----
  k_prep<<<256, 256, 0, stream>>>(warped_xyz, warped_points, f2_xyz, f2_points,
                                  lidar_z, wxyz, pin, qin, pinb, qiTb);
  k_f2bf<<<32, 256, 0, stream>>>(W1, w1b, 128 * 64);
  k_f2bf<<<16, 256, 0, stream>>>(W2, w2b, 64 * 64);
  k_f2bf<<<96, 256, 0, stream>>>(WA, w2ab, 192 * 128);
  k_f2bf<<<32, 256, 0, stream>>>(WB, w2bb, 128 * 64);
  k_hphq<<<1024, 256, 0, stream>>>(W0, wxyz, pin, qin, f2_xyz, hp, hq);
  k_bn1_stats<<<1, 128, 0, stream>>>(hp, hq, g0, a1, mup, muq);
  k_fold<<<512, 256, 0, stream>>>(hp, hq, a1, mup, muq, be0, hp2b, hq2b);
  k_sim<<<512, 256, 0, stream>>>(pinb, qiTb, sim);
  k_rowmax<<<128, 256, 0, stream>>>(sim, rmax);
  k_colmax<<<4, 256, 0, stream>>>(sim, cinv);

  // ---- pairwise MLP: 3 passes (BN-training stats) ----
  k_pair_mlp<0><<<128, 256, 0, stream>>>(hp2b, hq2b, w1b, w2b, b1, nullptr,
                                         nullptr, nullptr, nullptr, nullptr,
                                         nullptr, nullptr, nullptr, partS,
                                         partQ, nullptr);
  k_reduce_cols<<<1, 64, 0, stream>>>(partS, 1024, 64, st2S);
  k_reduce_cols<<<1, 64, 0, stream>>>(partQ, 1024, 64, st2Q);
  k_finalize_ao<<<1, 64, 0, stream>>>(st2S, st2Q, 1048576.f, g1, be1, a2, o2, 64);
  k_pair_mlp<1><<<128, 256, 0, stream>>>(hp2b, hq2b, w1b, w2b, b1, a2, o2, b2,
                                         nullptr, nullptr, nullptr, nullptr,
                                         nullptr, partS, partQ, nullptr);
  k_reduce_cols<<<1, 64, 0, stream>>>(partS, 1024, 64, st3S);
  k_reduce_cols<<<1, 64, 0, stream>>>(partQ, 1024, 64, st3Q);
  k_finalize_ao<<<1, 64, 0, stream>>>(st3S, st3Q, 1048576.f, g2, be2, a3, o3, 64);
  k_pair_mlp<2><<<128, 256, 0, stream>>>(hp2b, hq2b, w1b, w2b, b1, a2, o2, b2,
                                         a3, o3, sim, rmax, cinv, nullptr,
                                         nullptr, pifeat);

  // ---- KNN grouping + pc encoder + mlp2 ----
  k_knn<<<128, 256, 0, stream>>>(wxyz, idx);
  k_enc<<<64, 256, 0, stream>>>(wxyz, idx, We, bE, wsenc);
  k_stats1<<<64, 256, 0, stream>>>(wsenc, 16384, 64, partS, partQ);
  k_reduce_cols<<<1, 64, 0, stream>>>(partS, 256, 64, stES);
  k_reduce_cols<<<1, 64, 0, stream>>>(partQ, 256, 64, stEQ);
  k_finalize_ao<<<1, 64, 0, stream>>>(stES, stEQ, 16384.f, gE, beE, aE, oE, 64);
  k_mlp2a<<<128, 256, 0, stream>>>(wsenc, aE, oE, warped_points, pifeat, idx,
                                   w2ab, bA, wsa);
  k_stats1<<<64, 256, 0, stream>>>(wsa, 16384, 128, partS, partQ);
  k_reduce_cols<<<1, 128, 0, stream>>>(partS, 128, 128, stAS);
  k_reduce_cols<<<1, 128, 0, stream>>>(partQ, 128, 128, stAQ);
  k_finalize_ao<<<1, 128, 0, stream>>>(stAS, stAQ, 16384.f, gA, beA, aM1, oM1, 128);
  k_mlp2b<<<128, 256, 0, stream>>>(wsa, aM1, oM1, w2bb, bB, wsb);
  k_stats1<<<64, 256, 0, stream>>>(wsb, 16384, 64, partS, partQ);
  k_reduce_cols<<<1, 64, 0, stream>>>(partS, 256, 64, stBS);
  k_reduce_cols<<<1, 64, 0, stream>>>(partQ, 256, 64, stBQ);
  k_finalize_ao<<<1, 64, 0, stream>>>(stBS, stBQ, 16384.f, gB, beB, aM2, oM2, 64);
  k_softmax_out<<<128, 256, 0, stream>>>(wsb, aM2, oM2, idx, pifeat,
                                         (float*)d_out);
}
